// GNN_78486232367466
// MI455X (gfx1250) — matrix operand order repaired and checked by reading, not compiled, not run on hardware
//
#include <hip/hip_runtime.h>
#include <hip/hip_bf16.h>

typedef __attribute__((ext_vector_type(16))) __bf16 v16bf;
typedef __attribute__((ext_vector_type(8)))  __bf16 v8bf;
typedef __attribute__((ext_vector_type(8)))  float  v8f;
typedef __attribute__((ext_vector_type(4)))  float  v4f;
typedef const __attribute__((address_space(1))) v4f* gv4p;

#define GN_N 25600
#define GN_E 51200
#define GN_G 256
#define HID 256
#define BM  32
#define KC  128
#define APAD 8
#define ASTRIDE (KC + APAD)   // 136 halves = 272B row stride (16B-aligned, bank-shifted)

struct Seg {
    const float* p;   // base pointer [rows][width]
    const int*   idx; // gather index (nullable)
    int          div; // if idx==null: row = (div>1) ? R/div : R
    int          width;
};

__device__ __forceinline__ int seg_row(const Seg& s, int R) {
    if (s.idx) return s.idx[R];
    return (s.div > 1) ? (R / s.div) : R;
}

// Fused gather-concat GEMM (bf16 WMMA, f32 accum) + bias + LayerNorm + ReLU.
// Output: out[M][256]. Wt is bf16 [256][K] (pre-transposed weights).
__global__ void __launch_bounds__(256)
mlp_fused(Seg s0, Seg s1, Seg s2, Seg s3,
          const __bf16* __restrict__ Wt, int K,
          const float* __restrict__ bias,
          const float* __restrict__ gamma,
          const float* __restrict__ beta,
          float* __restrict__ out)
{
    __shared__ __align__(16) __bf16 smA[2][BM * ASTRIDE];  // ping-pong A tiles
    __shared__ float smO[BM * HID];
    __shared__ float smRed[2 * BM * 8];

    const int tid      = threadIdx.x;
    const int lane     = tid & 31;
    const int w        = tid >> 5;
    const int rowBase  = (w >> 2) * 16;          // 0 or 16
    const int nBase    = (w & 3) * 64;           // 0,64,128,192
    const int blockRow0 = blockIdx.x * BM;

    const int c1 = s0.width;
    const int c2 = c1 + s1.width;
    const int c3 = c2 + s2.width;

    // staging mapping: 256 threads cover 32 rows x 128 cols (16 elems each)
    const int sr = tid >> 3;          // 0..31
    const int sc = (tid & 7) * 16;    // 0..112

    const int R  = blockRow0 + sr;
    // Adjusted per-thread row addresses: global address = qX + 4*gk.
    const uintptr_t q0 = (uintptr_t)(s0.p + (size_t)seg_row(s0, R) * s0.width);
    const uintptr_t q1 = (uintptr_t)(s1.p + (size_t)seg_row(s1, R) * s1.width) - (uintptr_t)c1 * 4u;
    const uintptr_t q2 = (uintptr_t)(s2.p + (size_t)seg_row(s2, R) * s2.width) - (uintptr_t)c2 * 4u;
    const uintptr_t q3 = (uintptr_t)(s3.p + (size_t)seg_row(s3, R) * s3.width) - (uintptr_t)c3 * 4u;

    v8f acc[4];
    #pragma unroll
    for (int t = 0; t < 4; t++)
        #pragma unroll
        for (int i = 0; i < 8; i++) acc[t][i] = 0.0f;

    const int aRow  = rowBase + (lane & 15);
    const int aKsel = (lane < 16) ? 0 : 16;
    const int bKsel = (lane < 16) ? 0 : 16;
    const int aOff  = aRow * ASTRIDE + aKsel;

    // Per-lane weight pointer: lane (n = nBase + lane&15) reads contiguous 32B
    // per k-step; tile t adds t*16 rows of K elements.
    const __bf16* wtLane = Wt + (size_t)(nBase + (lane & 15)) * K + bKsel;

    // Staging registers (chunk-level global->reg->LDS pipeline).
    v4f f0, f1, f2, f3;
    auto stage_load = [&](int kc) {
        const int gk = kc + sc;
        uintptr_t sp = (gk < c1) ? q0 : (gk < c2) ? q1 : (gk < c3) ? q2 : q3;
        gv4p v = (gv4p)(sp + (uintptr_t)gk * 4u);   // addrspace(1): global_load_b128
        f0 = v[0]; f1 = v[1]; f2 = v[2]; f3 = v[3];
    };
    auto stage_store = [&](__bf16* buf) {
        v8bf p0, p1;
        p0[0]=(__bf16)f0[0]; p0[1]=(__bf16)f0[1]; p0[2]=(__bf16)f0[2]; p0[3]=(__bf16)f0[3];
        p0[4]=(__bf16)f1[0]; p0[5]=(__bf16)f1[1]; p0[6]=(__bf16)f1[2]; p0[7]=(__bf16)f1[3];
        p1[0]=(__bf16)f2[0]; p1[1]=(__bf16)f2[1]; p1[2]=(__bf16)f2[2]; p1[3]=(__bf16)f2[3];
        p1[4]=(__bf16)f3[0]; p1[5]=(__bf16)f3[1]; p1[6]=(__bf16)f3[2]; p1[7]=(__bf16)f3[3];
        __bf16* dst = buf + sr * ASTRIDE + sc;
        *(v8bf*)(dst)     = p0;
        *(v8bf*)(dst + 8) = p1;
    };

    auto loadB = [&](v16bf (&buf)[4], int kabs) {
        #pragma unroll
        for (int t = 0; t < 4; t++)
            buf[t] = *(const v16bf*)(wtLane + (size_t)t * 16 * K + kabs);
    };
    auto loadA = [&](const __bf16* aB, int k0) -> v16bf {
        const __bf16* ap = aB + k0;
        v8bf lo = *(const v8bf*)ap;
        v8bf hi = *(const v8bf*)(ap + 8);
        return __builtin_shufflevector(lo, hi,
            0,1,2,3,4,5,6,7,8,9,10,11,12,13,14,15);
    };
    auto mma = [&](v16bf a, const v16bf (&buf)[4]) {
        #pragma unroll
        for (int t = 0; t < 4; t++)
            acc[t] = __builtin_amdgcn_wmma_f32_16x16x32_bf16(
                false, a, false, buf[t], (short)0, acc[t], false, false);
    };

    // Prologue: stage chunk 0, prefetch step-0 B.
    v16bf bb0[4], bb1[4];
    stage_load(0);
    stage_store(smA[0]);
    loadB(bb0, 0);
    __syncthreads();

    int cur = 0;
    for (int kc = 0; kc < K; kc += KC) {
        const __bf16* aB = &smA[cur][aOff];
        const bool haveNext = (kc + KC < K);          // block-uniform
        if (haveNext) stage_load(kc + KC);            // next chunk's gathers in flight
        __builtin_amdgcn_sched_barrier(0);
        // ---- 4 k-steps, fully unrolled; prefetch(s+1) fenced before mma(s) ----
        loadB(bb1, kc + 32);
        v16bf a0 = loadA(aB, 0);
        v16bf a1 = loadA(aB, 32);
        __builtin_amdgcn_sched_barrier(0);
        mma(a0, bb0);                                 // step 0
        __builtin_amdgcn_sched_barrier(0);
        loadB(bb0, kc + 64);
        a0 = loadA(aB, 64);
        __builtin_amdgcn_sched_barrier(0);
        mma(a1, bb1);                                 // step 1
        __builtin_amdgcn_sched_barrier(0);
        loadB(bb1, kc + 96);
        a1 = loadA(aB, 96);
        __builtin_amdgcn_sched_barrier(0);
        mma(a0, bb0);                                 // step 2
        __builtin_amdgcn_sched_barrier(0);
        if (haveNext) loadB(bb0, kc + KC);            // next chunk step-0 B
        __builtin_amdgcn_sched_barrier(0);
        mma(a1, bb1);                                 // step 3
        if (haveNext) stage_store(smA[cur ^ 1]);      // cvt + 2x ds_store_b128
        __syncthreads();
        cur ^= 1;
    }

    // ---- spill accumulators to LDS (C/D layout: VGPR v -> M=v (+8 for hi lanes)) ----
    {
        const int dRow = rowBase + ((lane >= 16) ? 8 : 0);
        #pragma unroll
        for (int t = 0; t < 4; t++) {
            const int col = nBase + t*16 + (lane & 15);
            #pragma unroll
            for (int v = 0; v < 8; v++)
                smO[(dRow + v) * HID + col] = acc[t][v];
        }
    }
    __syncthreads();

    // ---- bias + LayerNorm + ReLU: 8 threads per row ----
    {
        const int r  = tid >> 3;
        const int cs = (tid & 7) * 32;
        float sum = 0.f, sq = 0.f;
        #pragma unroll 8
        for (int j = 0; j < 32; j++) {
            float v = smO[r * HID + cs + j] + bias[cs + j];
            smO[r * HID + cs + j] = v;
            sum += v; sq += v * v;
        }
        smRed[r * 8 + (tid & 7)]            = sum;
        smRed[BM * 8 + r * 8 + (tid & 7)]   = sq;
        __syncthreads();
        float ts = 0.f, tq = 0.f;
        #pragma unroll
        for (int j = 0; j < 8; j++) {
            ts += smRed[r * 8 + j];
            tq += smRed[BM * 8 + r * 8 + j];
        }
        const float mu   = ts * (1.0f / HID);
        const float var  = tq * (1.0f / HID) - mu * mu;
        const float rstd = rsqrtf(var + 1e-5f);
        float* orow = out + (size_t)(blockRow0 + r) * HID + cs;
        #pragma unroll 8
        for (int j = 0; j < 32; j++) {
            float v = (smO[r * HID + cs + j] - mu) * rstd * gamma[cs + j] + beta[cs + j];
            orow[j] = fmaxf(v, 0.0f);
        }
    }
}

// transpose f32 [K][Nn] -> bf16 [Nn][K]
__global__ void transpose_to_bf16(const float* __restrict__ src,
                                  __bf16* __restrict__ dst, int K, int Nn)
{
    int t = blockIdx.x * 256 + threadIdx.x;
    if (t >= K * Nn) return;
    int k = t / Nn, n = t - k * Nn;
    dst[(size_t)n * K + k] = (__bf16)src[t];
}

// out[M][256] = x[M][Kin] @ W[Kin][256] + b
__global__ void embed_kernel(const float* __restrict__ x,
                             const float* __restrict__ W,
                             const float* __restrict__ b,
                             float* __restrict__ out, int Kin)
{
    int c = threadIdx.x, m = blockIdx.x;
    const float* xr = x + (size_t)m * Kin;
    float s = b[c];
    for (int k = 0; k < Kin; k++) s = fmaf(xr[k], W[k * 256 + c], s);
    out[(size_t)m * 256 + c] = s;
}

__global__ void scatter_add(const float* __restrict__ edges,
                            const int* __restrict__ snd,
                            const int* __restrict__ rcv,
                            float* __restrict__ sagg, float* __restrict__ ragg)
{
    int e = blockIdx.x, c = threadIdx.x;
    float v = edges[(size_t)e * 256 + c];
    int s = snd[e], r = rcv[e];
    unsafeAtomicAdd(&sagg[(size_t)s * 256 + c], v);
    unsafeAtomicAdd(&ragg[(size_t)r * 256 + c], v);
}

__global__ void graph_agg(const float* __restrict__ nodes,
                          const float* __restrict__ edges,
                          float* __restrict__ nagg, float* __restrict__ eagg)
{
    int g = blockIdx.x, c = threadIdx.x;
    float s = 0.f;
    const float* p = nodes + (size_t)g * 100 * 256 + c;
    for (int i = 0; i < 100; i++) s += p[(size_t)i * 256];
    nagg[(size_t)g * 256 + c] = s;
    float s2 = 0.f;
    const float* q = edges + (size_t)g * 200 * 256 + c;
    for (int i = 0; i < 200; i++) s2 += q[(size_t)i * 256];
    eagg[(size_t)g * 256 + c] = s2;
}

__global__ void decoder_kernel(const float* __restrict__ glob,
                               const float* __restrict__ dW,
                               const float* __restrict__ db,
                               float* __restrict__ out)
{
    int g = blockIdx.x, o = threadIdx.x; // 128 threads
    float s = db[o];
    const float* gr = glob + (size_t)g * 256;
    for (int k = 0; k < 256; k++) s = fmaf(gr[k], dW[k * 128 + o], s);
    out[(size_t)g * 128 + o] = s;
}

extern "C" void kernel_launch(void* const* d_in, const int* in_sizes, int n_in,
                              void* d_out, int out_size, void* d_ws, size_t ws_size,
                              hipStream_t stream)
{
    const float* nodes     = (const float*)d_in[0];
    const float* edges     = (const float*)d_in[1];
    const int*   senders   = (const int*)d_in[2];
    const int*   receivers = (const int*)d_in[3];
    const float* ne_W = (const float*)d_in[6];  const float* ne_b = (const float*)d_in[7];
    const float* ee_W = (const float*)d_in[8];  const float* ee_b = (const float*)d_in[9];
    const float* eW0 = (const float*)d_in[10]; const float* eb0 = (const float*)d_in[11];
    const float* eg0 = (const float*)d_in[12]; const float* et0 = (const float*)d_in[13];
    const float* nW0 = (const float*)d_in[14]; const float* nb0 = (const float*)d_in[15];
    const float* ng0 = (const float*)d_in[16]; const float* nt0 = (const float*)d_in[17];
    const float* gW0 = (const float*)d_in[18]; const float* gb0 = (const float*)d_in[19];
    const float* gg0 = (const float*)d_in[20]; const float* gt0 = (const float*)d_in[21];
    const float* eWs = (const float*)d_in[22]; const float* ebs = (const float*)d_in[23];
    const float* egs = (const float*)d_in[24]; const float* ets = (const float*)d_in[25];
    const float* nWs = (const float*)d_in[26]; const float* nbs = (const float*)d_in[27];
    const float* ngs = (const float*)d_in[28]; const float* nts = (const float*)d_in[29];
    const float* gWs = (const float*)d_in[30]; const float* gbs = (const float*)d_in[31];
    const float* ggs = (const float*)d_in[32]; const float* gts = (const float*)d_in[33];
    const float* dW  = (const float*)d_in[34]; const float* db  = (const float*)d_in[35];

    // ---- workspace carve-out ----
    char* ws = (char*)d_ws;
    size_t off = 0;
    auto take = [&](size_t bytes) -> char* {
        char* r = ws + off;
        off += (bytes + 255) & ~(size_t)255;
        return r;
    };
    const size_t F = sizeof(float);
    float* h_nodes = (float*)take((size_t)GN_N * 256 * F);
    float* h_edges = (float*)take((size_t)GN_E * 256 * F);
    float* s_agg   = (float*)take((size_t)GN_N * 256 * F);
    float* r_agg   = (float*)take((size_t)GN_N * 256 * F);
    float* globA   = (float*)take((size_t)GN_G * 256 * F);
    float* globB   = (float*)take((size_t)GN_G * 256 * F);
    float* n_agg   = (float*)take((size_t)GN_G * 256 * F);
    float* e_agg   = (float*)take((size_t)GN_G * 256 * F);
    __bf16* eW0t = (__bf16*)take((size_t)256 * 896 * 2);
    __bf16* nW0t = (__bf16*)take((size_t)256 * 896 * 2);
    __bf16* gW0t = (__bf16*)take((size_t)256 * 640 * 2);
    __bf16* eWt  = (__bf16*)take((size_t)4 * 256 * 1024 * 2);
    __bf16* nWt  = (__bf16*)take((size_t)4 * 256 * 1024 * 2);
    __bf16* gWt  = (__bf16*)take((size_t)4 * 256 * 768 * 2);

    // ---- weight prep: transpose + bf16 ----
    auto tr = [&](const float* src, __bf16* dst, int K, int Nn) {
        int tot = K * Nn;
        transpose_to_bf16<<<(tot + 255) / 256, 256, 0, stream>>>(src, dst, K, Nn);
    };
    tr(eW0, eW0t, 896, 256);
    tr(nW0, nW0t, 896, 256);
    tr(gW0, gW0t, 640, 256);
    for (int i = 0; i < 4; i++) {
        tr(eWs + (size_t)i * 1024 * 256, eWt + (size_t)i * 256 * 1024, 1024, 256);
        tr(nWs + (size_t)i * 1024 * 256, nWt + (size_t)i * 256 * 1024, 1024, 256);
        tr(gWs + (size_t)i * 768 * 256,  gWt + (size_t)i * 256 * 768,  768, 256);
    }

    // ---- embedders + initial globals ----
    hipMemsetAsync(globA, 0, (size_t)GN_G * 128 * F, stream);
    embed_kernel<<<GN_N, 256, 0, stream>>>(nodes, ne_W, ne_b, h_nodes, 9);
    embed_kernel<<<GN_E, 256, 0, stream>>>(edges, ee_W, ee_b, h_edges, 3);

    // ---- message-passing steps ----
    float* gin = globA; int gw = 128;
    for (int s = 0; s < 5; s++) {
        const __bf16 *ewt, *nwt, *gwt;
        const float *eb_, *eg_, *et_, *nb_, *ng_, *nt_, *gb_, *gg_, *gt_;
        if (s == 0) {
            ewt = eW0t; eb_ = eb0; eg_ = eg0; et_ = et0;
            nwt = nW0t; nb_ = nb0; ng_ = ng0; nt_ = nt0;
            gwt = gW0t; gb_ = gb0; gg_ = gg0; gt_ = gt0;
        } else {
            int i = s - 1;
            ewt = eWt + (size_t)i * 256 * 1024; eb_ = ebs + i*256; eg_ = egs + i*256; et_ = ets + i*256;
            nwt = nWt + (size_t)i * 256 * 1024; nb_ = nbs + i*256; ng_ = ngs + i*256; nt_ = nts + i*256;
            gwt = gWt + (size_t)i * 256 * 768;  gb_ = gbs + i*256; gg_ = ggs + i*256; gt_ = gts + i*256;
        }
        const int Ke = 768 + gw;   // edge/node MLP input width (896 or 1024)
        const int Kg = 512 + gw;   // global MLP input width (640 or 768)

        // edge MLP: [h_edges | h_nodes[snd] | h_nodes[rcv] | globals] -> h_edges
        {
            Seg s0{h_edges, nullptr, 1, 256};
            Seg s1{h_nodes, senders, 0, 256};
            Seg s2{h_nodes, receivers, 0, 256};
            Seg s3{gin, nullptr, 200, gw};
            mlp_fused<<<GN_E / BM, 256, 0, stream>>>(s0, s1, s2, s3, ewt, Ke,
                                                     eb_, eg_, et_, h_edges);
        }
        // segment sums of new edges into per-node aggregates
        hipMemsetAsync(s_agg, 0, (size_t)GN_N * 256 * F, stream);
        hipMemsetAsync(r_agg, 0, (size_t)GN_N * 256 * F, stream);
        scatter_add<<<GN_E, 256, 0, stream>>>(h_edges, senders, receivers, s_agg, r_agg);
        // node MLP: [h_nodes | s_agg | r_agg | globals] -> h_nodes
        {
            Seg s0{h_nodes, nullptr, 1, 256};
            Seg s1{s_agg,   nullptr, 1, 256};
            Seg s2{r_agg,   nullptr, 1, 256};
            Seg s3{gin, nullptr, 100, gw};
            mlp_fused<<<GN_N / BM, 256, 0, stream>>>(s0, s1, s2, s3, nwt, Ke,
                                                     nb_, ng_, nt_, h_nodes);
        }
        // per-graph reductions (contiguous rows)
        graph_agg<<<GN_G, 256, 0, stream>>>(h_nodes, h_edges, n_agg, e_agg);
        // global MLP: [n_agg | e_agg | globals] -> globals (ping-pong)
        float* gout = (gin == globA) ? globB : globA;
        {
            Seg s0{n_agg, nullptr, 1, 256};
            Seg s1{e_agg, nullptr, 1, 256};
            Seg s2{gin,   nullptr, 1, gw};
            Seg s3{gin,   nullptr, 1, 0};   // unused (K == b3)
            mlp_fused<<<GN_G / BM, 256, 0, stream>>>(s0, s1, s2, s3, gwt, Kg,
                                                     gb_, gg_, gt_, gout);
        }
        gin = gout; gw = 256;
    }

    // ---- decoder ----
    decoder_kernel<<<GN_G, 128, 0, stream>>>(gin, dW, db, (float*)d_out);
}
